// AttentionHead_1949915152457
// MI455X (gfx1250) — matrix operand order repaired and checked by reading, not compiled, not run on hardware
//
#include <hip/hip_runtime.h>
#include <hip/hip_bf16.h>
#include <stdint.h>
#include <cmath>

// Problem constants
#define BB   8
#define SS   4096
#define DIN  1024
#define DD   128
#define NJ   (SS / 32)

typedef __attribute__((ext_vector_type(16))) __bf16 v16bf;
typedef __attribute__((ext_vector_type(8)))  float  v8f;

union FragB { v16bf v; uint32_t u[8]; };

// Hardware f32 -> bf16 (RNE) conversion; backend selects v_cvt_*bf16_f32.
__device__ __forceinline__ uint16_t bf16bits(float f) {
  union { __bf16 b; uint16_t s; } y;
  y.b = (__bf16)f;
  return y.s;
}
__device__ __forceinline__ uint32_t pack_bf16(float lo, float hi) {
  union { __bf16 b[2]; uint32_t u; } y;
  y.b[0] = (__bf16)lo;
  y.b[1] = (__bf16)hi;
  return y.u;
}

__device__ __forceinline__ v8f wmma_bf16(v16bf a, v16bf b, v8f c) {
  return __builtin_amdgcn_wmma_f32_16x16x32_bf16(false, a, false, b, (short)0, c,
                                                 false, false);
}

// ---- CDNA5 async global->LDS copy (ASYNCcnt-tracked) ------------------------
#if __has_builtin(__builtin_amdgcn_global_load_async_to_lds_b32)
typedef __attribute__((address_space(1))) int as1_int;
typedef __attribute__((address_space(3))) int as3_int;
__device__ __forceinline__ void async_copy_b32(const uint32_t* g, uint32_t* l) {
  __builtin_amdgcn_global_load_async_to_lds_b32((as1_int*)(void*)g,
                                                (as3_int*)(void*)l, 0, 0);
}
#else
__device__ __forceinline__ void async_copy_b32(const uint32_t* g, uint32_t* l) {
  uint32_t laddr = (uint32_t)(uintptr_t)l;   // low 32 bits of generic = LDS offset
  asm volatile("global_load_async_to_lds_b32 %0, %1, off"
               :: "v"(laddr), "v"(g) : "memory");
}
#endif

__device__ __forceinline__ void wait_async0() {
#if __has_builtin(__builtin_amdgcn_s_wait_asynccnt)
  __builtin_amdgcn_s_wait_asynccnt(0);
#else
  asm volatile("s_wait_asynccnt 0" ::: "memory");
#endif
}

// A-fragment (16x32 bf16) from row-major bf16 buffer p[row*stride + k].
// Contiguous k order (this code's choice): lane L: m=L&15, half=L>>4; VGPR i: k0=half*16+2*i
__device__ __forceinline__ v16bf load_a_frag(const uint16_t* p, int stride, int lane) {
  const int m = lane & 15, h = lane >> 4;
  const uint32_t* q = (const uint32_t*)p;
  const int rs = stride >> 1;
  FragB f;
#pragma unroll
  for (int i = 0; i < 8; ++i) {
    int k0 = h * 16 + (i << 1);
    f.u[i] = q[m * rs + (k0 >> 1)];
  }
  return f.v;
}

// B-fragment (32x16 bf16) where B(k,n) = p[n*stride + k].
// Contiguous k order (this code's choice): lane L: n=L&15, half=L>>4; VGPR i: k = half*16 + 2i, 2i+1.
__device__ __forceinline__ v16bf load_bT_frag(const uint16_t* p, int stride, int lane) {
  const int n = lane & 15, h = lane >> 4;
  const uint32_t* q = (const uint32_t*)p;
  const int rs = stride >> 1;
  FragB f;
#pragma unroll
  for (int i = 0; i < 8; ++i) f.u[i] = q[n * rs + h * 8 + i];
  return f.v;
}

// ---------------- Projection: Y = bf16( (X * W + bias) * outscale ) ----------
__global__ __launch_bounds__(256) void proj_kernel(const float* __restrict__ X,
                                                   const float* __restrict__ W,
                                                   const float* __restrict__ bias,
                                                   uint16_t* __restrict__ Y,
                                                   float outscale) {
  const int tid = threadIdx.x;
  const int wave = tid >> 5, lane = tid & 31;
  const int halfg = lane >> 4, ln = lane & 15;
  const int rowbase = blockIdx.x * 128;

  __shared__ __align__(16) uint16_t Xc[128 * 34];  // bf16 X chunk  [row][k]
  __shared__ __align__(16) uint16_t WT[128 * 34];  // bf16 W chunk  [col][k]

  v8f acc[8] = {};

  for (int kc = 0; kc < DIN / 32; ++kc) {
    __syncthreads();
    // stage X chunk: 128 rows x 32 k, float2 -> packed bf16 dword
    for (int i = tid; i < 128 * 16; i += 256) {
      int r = i >> 4, c = i & 15;                  // cols 2c, 2c+1
      float2 xy = ((const float2*)(X + (size_t)(rowbase + r) * DIN + kc * 32))[c];
      ((uint32_t*)Xc)[r * 17 + c] = pack_bf16(xy.x, xy.y);
    }
    // stage W^T chunk: [col][k]; float2 over cols, 16-bit scatter (transpose)
    for (int i = tid; i < 32 * 64; i += 256) {
      int k = i >> 6, c2 = i & 63;                 // cols 2c2, 2c2+1
      float2 w2 = ((const float2*)(W + (size_t)(kc * 32 + k) * DD))[c2];
      WT[(2 * c2) * 34 + k]     = bf16bits(w2.x);
      WT[(2 * c2 + 1) * 34 + k] = bf16bits(w2.y);
    }
    __syncthreads();

    v16bf a = load_a_frag(Xc + wave * 16 * 34, 34, lane);
#pragma unroll
    for (int t = 0; t < 8; ++t) {
      v16bf bw = load_bT_frag(WT + t * 16 * 34, 34, lane);
      acc[t] = wmma_bf16(a, bw, acc[t]);
    }
  }

#pragma unroll
  for (int t = 0; t < 8; ++t) {
    int col = t * 16 + ln;
    float bi = bias[col];
#pragma unroll
    for (int r = 0; r < 8; ++r) {
      int row = rowbase + wave * 16 + r + 8 * halfg;
      Y[(size_t)row * DD + col] = bf16bits((acc[t][r] + bi) * outscale);
    }
  }
}

// ---------------- Flash attention core (double-buffered, async staging) ------
__global__ __launch_bounds__(256) void flash_kernel(const uint16_t* __restrict__ Qb,
                                                    const uint16_t* __restrict__ Kb,
                                                    const uint16_t* __restrict__ Vb,
                                                    float* __restrict__ Out) {
  const int tid = threadIdx.x;
  const int wave = tid >> 5, lane = tid & 31;
  const int halfg = lane >> 4, ln = lane & 15;
  const int b = blockIdx.y;
  const int qbase = blockIdx.x * 128 + wave * 16;

  __shared__ __align__(16) uint16_t Klds[2][32 * 130];   // [key][dim], stride 130
  __shared__ __align__(16) uint16_t Vt[2][128 * 34];     // [dim][key], stride 34
  __shared__ __align__(16) uint16_t Plds[8 * 16 * 34];   // per-wave P, stride 34

  // Q A-fragments (loaded once from global; Q pre-scaled by 1/sqrt(128))
  v16bf aq[4];
  {
    const uint32_t* qg = (const uint32_t*)Qb;
    const size_t rowdw = ((size_t)b * SS + qbase + ln) * (DD / 2);
#pragma unroll
    for (int kc = 0; kc < 4; ++kc) {
      FragB f;
#pragma unroll
      for (int i = 0; i < 8; ++i) {
        int k0 = kc * 32 + halfg * 16 + (i << 1);
        f.u[i] = qg[rowdw + (k0 >> 1)];
      }
      aq[kc] = f.v;
    }
  }

  // staging: K via async global->LDS dword copies; V transposed via VGPR path
  auto stage = [&](int j, int buf) {
    const uint32_t* kg = (const uint32_t*)Kb + ((size_t)b * SS + (size_t)j * 32) * 64;
    uint32_t* kl = (uint32_t*)Klds[buf];
    for (int i = tid; i < 32 * 64; i += 256) {
      int key = i >> 6, dp = i & 63;
      async_copy_b32(kg + key * 64 + dp, kl + key * 65 + dp);
    }
    const uint32_t* vg = (const uint32_t*)Vb + ((size_t)b * SS + (size_t)j * 32) * 64;
    uint16_t* vt = Vt[buf];
    for (int i = tid; i < 32 * 64; i += 256) {
      int key = i >> 6, dp = i & 63;               // dims 2dp, 2dp+1 of row `key`
      uint32_t w = vg[key * 64 + dp];
      vt[(2 * dp) * 34 + key] = (uint16_t)w;
      vt[(2 * dp + 1) * 34 + key] = (uint16_t)(w >> 16);
    }
  };

  float m_r[8], l_r[8];
#pragma unroll
  for (int r = 0; r < 8; ++r) { m_r[r] = -3.0e38f; l_r[r] = 0.f; }
  v8f acc[8] = {};

  stage(0, 0);
  for (int j = 0; j < NJ; ++j) {
    const int buf = j & 1;
    wait_async0();                 // my async LDS writes for buf are done
    __syncthreads();               // everyone's staging of buf is visible
    if (j + 1 < NJ) stage(j + 1, buf ^ 1);
    if (j + 2 < NJ) {              // prefetch tile j+2 toward L2/WGP$
      const char* pk = (const char*)(Kb + ((size_t)b * SS + (size_t)(j + 2) * 32) * DD);
      const char* pv = (const char*)(Vb + ((size_t)b * SS + (size_t)(j + 2) * 32) * DD);
      __builtin_prefetch(pk + tid * 32, 0, 0);
      __builtin_prefetch(pv + tid * 32, 0, 0);
    }

    // ---- scores: S = Q * K^T (two 16x16 tiles over 32 keys) ----
    v8f s0 = {}, s1 = {};
    const uint32_t* kl = (const uint32_t*)Klds[buf];
#pragma unroll
    for (int kc = 0; kc < 4; ++kc) {
      FragB f0, f1;
#pragma unroll
      for (int i = 0; i < 8; ++i) {
        int dw = kc * 16 + halfg * 8 + i;
        f0.u[i] = kl[ln * 65 + dw];
        f1.u[i] = kl[(16 + ln) * 65 + dw];
      }
      s0 = wmma_bf16(aq[kc], f0.v, s0);
      s1 = wmma_bf16(aq[kc], f1.v, s1);
    }

    // ---- online softmax (row spans 16 lanes of a half-wave) ----
    float rmax[8];
#pragma unroll
    for (int r = 0; r < 8; ++r) rmax[r] = fmaxf(s0[r], s1[r]);
#pragma unroll
    for (int d = 1; d < 16; d <<= 1) {
#pragma unroll
      for (int r = 0; r < 8; ++r) rmax[r] = fmaxf(rmax[r], __shfl_xor(rmax[r], d));
    }
    float corr[8], p0[8], p1[8], rsum[8];
#pragma unroll
    for (int r = 0; r < 8; ++r) {
      float mn = fmaxf(m_r[r], rmax[r]);
      corr[r] = __expf(m_r[r] - mn);
      m_r[r] = mn;
      p0[r] = __expf(s0[r] - mn);
      p1[r] = __expf(s1[r] - mn);
      rsum[r] = p0[r] + p1[r];
    }
#pragma unroll
    for (int d = 1; d < 16; d <<= 1) {
#pragma unroll
      for (int r = 0; r < 8; ++r) rsum[r] += __shfl_xor(rsum[r], d);
    }
#pragma unroll
    for (int r = 0; r < 8; ++r) l_r[r] = l_r[r] * corr[r] + rsum[r];
#pragma unroll
    for (int t = 0; t < 8; ++t) {
#pragma unroll
      for (int r = 0; r < 8; ++r) acc[t][r] = acc[t][r] * corr[r];
    }

    // ---- P (C layout) -> LDS -> A layout (wave-private region) ----
    uint16_t* Pw = Plds + wave * 16 * 34;
#pragma unroll
    for (int r = 0; r < 8; ++r) {
      int row = r + 8 * halfg;
      Pw[row * 34 + ln] = bf16bits(p0[r]);
      Pw[row * 34 + 16 + ln] = bf16bits(p1[r]);
    }
    asm volatile("s_wait_dscnt 0" ::: "memory");
    v16bf ap = load_a_frag(Pw, 34, lane);

    // ---- O += P * V ----
    const uint32_t* vt = (const uint32_t*)Vt[buf];
#pragma unroll
    for (int t = 0; t < 8; ++t) {
      FragB f;
#pragma unroll
      for (int i = 0; i < 8; ++i) f.u[i] = vt[(t * 16 + ln) * 17 + halfg * 8 + i];
      acc[t] = wmma_bf16(ap, f.v, acc[t]);
    }
  }

  // ---- epilogue: O / l ----
#pragma unroll
  for (int r = 0; r < 8; ++r) {
    float inv = 1.0f / l_r[r];
    int row = qbase + r + 8 * halfg;
    float* op = Out + ((size_t)b * SS + row) * DD + ln;
#pragma unroll
    for (int t = 0; t < 8; ++t) op[t * 16] = acc[t][r] * inv;
  }
}

extern "C" void kernel_launch(void* const* d_in, const int* in_sizes, int n_in,
                              void* d_out, int out_size, void* d_ws, size_t ws_size,
                              hipStream_t stream) {
  (void)in_sizes; (void)n_in; (void)out_size; (void)ws_size;
  const float* query = (const float*)d_in[0];
  const float* key   = (const float*)d_in[1];
  const float* value = (const float*)d_in[2];
  const float* Wq    = (const float*)d_in[3];
  const float* bq    = (const float*)d_in[4];
  const float* Wk    = (const float*)d_in[5];
  const float* bk    = (const float*)d_in[6];
  const float* Wv    = (const float*)d_in[7];
  const float* bv    = (const float*)d_in[8];
  float* out = (float*)d_out;

  uint16_t* Qb = (uint16_t*)d_ws;                         // [B,S,128] bf16
  uint16_t* Kb = Qb + (size_t)BB * SS * DD;
  uint16_t* Vb = Kb + (size_t)BB * SS * DD;

  const float scale = 1.0f / sqrtf((float)DD);            // folded into Q
  dim3 blk(256);
  dim3 gproj((BB * SS) / 128);
  proj_kernel<<<gproj, blk, 0, stream>>>(query, Wq, bq, Qb, scale);
  proj_kernel<<<gproj, blk, 0, stream>>>(key,   Wk, bk, Kb, 1.0f);
  proj_kernel<<<gproj, blk, 0, stream>>>(value, Wv, bv, Vb, 1.0f);

  dim3 gflash(SS / 128, BB);
  flash_kernel<<<gflash, blk, 0, stream>>>(Qb, Kb, Vb, out);
}
